// CompressedModel_59433757442491
// MI455X (gfx1250) — matrix operand order repaired and checked by reading, not compiled, not run on hardware
//
#include <hip/hip_runtime.h>
#include <hip/hip_bf16.h>
#include <cstddef>
#include <cstdint>

#define B_     32
#define T_     2048
#define C_     768
#define R_     102
#define TWO_R_ 204
#define KEEP_  1844      // T - 2r
#define OUT_T_ 1946      // KEEP + r
#define MARGIN_ 0.5f

typedef __attribute__((ext_vector_type(16))) __bf16 v16bf;
typedef __attribute__((ext_vector_type(8)))  __bf16 v8bf;
typedef __attribute__((ext_vector_type(8)))  float  v8f;

union FragAB { v16bf v; v8bf h[2]; };

__device__ __forceinline__ unsigned short f32_to_bf16_bits(float f) {
  unsigned int u = __float_as_uint(f);
  u += 0x7FFFu + ((u >> 16) & 1u);           // round-to-nearest-even
  return (unsigned short)(u >> 16);
}

// CDNA5 async global->LDS copy (tracked by ASYNCcnt), 16B per lane.
__device__ __forceinline__ void async_ld_b128(unsigned lds_off, unsigned long long gaddr) {
  asm volatile("global_load_async_to_lds_b128 %0, %1, off"
               :: "v"(lds_off), "v"(gaddr) : "memory");
}
__device__ __forceinline__ void wait_async0() {
  asm volatile("s_wait_asynccnt 0" ::: "memory");
}

// ---------------------------------------------------------------- zero
__global__ void k_zero(float* __restrict__ p, int n) {
  int i = blockIdx.x * blockDim.x + threadIdx.x;
  if (i < n) p[i] = 0.f;
}

// ---------------------------------------------------------------- normalize + std
__global__ void __launch_bounds__(256) k_normalize(const float* __restrict__ x,
                                                   unsigned short* __restrict__ xn,
                                                   float* __restrict__ stdv) {
  const int tok = blockIdx.x;
  const int tid = threadIdx.x;
  const float* row = x + (size_t)tok * C_;
  const float v0 = row[tid], v1 = row[tid + 256], v2 = row[tid + 512];
  __shared__ float r1[256], r2[256];
  r1[tid] = v0 + v1 + v2;
  r2[tid] = v0 * v0 + v1 * v1 + v2 * v2;
  __syncthreads();
  for (int off = 128; off > 0; off >>= 1) {
    if (tid < off) { r1[tid] += r1[tid + off]; r2[tid] += r2[tid + off]; }
    __syncthreads();
  }
  const float sum = r1[0], sumsq = r2[0];
  const float inv = 1.f / sqrtf(sumsq);
  // std(xn, ddof=1): sum(xn^2)==1 exactly, so var = (1 - C*mean^2)/(C-1)
  const float mean = sum * inv / (float)C_;
  float var = (1.f - (float)C_ * mean * mean) / (float)(C_ - 1);
  var = fmaxf(var, 0.f);
  if (tid == 0) stdv[tok] = sqrtf(var);
  unsigned short* orow = xn + (size_t)tok * C_;
  orow[tid]       = f32_to_bf16_bits(v0 * inv);
  orow[tid + 256] = f32_to_bf16_bits(v1 * inv);
  orow[tid + 512] = f32_to_bf16_bits(v2 * inv);
}

// ---------------------------------------------------------------- Gram + col reduce
// 64(t) x 128(s) tile per 256-thread block; 8 waves as 4(row) x 2(col),
// each wave owns a 16x64 strip -> four v_wmma_f32_16x16x32_bf16 per K-step.
// Double-buffered LDS filled with GLOBAL_LOAD_ASYNC_TO_LDS_B128 (ASYNCcnt).
__global__ void __launch_bounds__(256) k_gram_colsum(const unsigned short* __restrict__ xn,
                                                     const float* __restrict__ stdv,
                                                     float* __restrict__ col_sum) {
  const int b   = blockIdx.z;
  const int t0  = blockIdx.y * 64;
  const int c0  = blockIdx.x * 128;
  const int tid  = threadIdx.x;
  const int lane = tid & 31;
  const int wave = tid >> 5;
  const int wrow = wave & 3;     // 0..3  -> A rows 16*wrow
  const int wcol = wave >> 2;    // 0..1  -> B cols 64*wcol
  const int m    = lane & 15;
  const int hi   = lane >> 4;

  __shared__ __align__(16) unsigned short lds_a[2][64 * 40];   // stride 40: 80B rows, 16B aligned
  __shared__ __align__(16) unsigned short lds_b[2][128 * 40];
  __shared__ float lds_std[64];
  __shared__ float lds_col[128];

  if (tid < 64)  lds_std[tid] = stdv[(size_t)b * T_ + t0 + tid];
  if (tid < 128) lds_col[tid] = 0.f;

  // loader mapping: A = 64 rows x 4 segs (1 per thread); B = 128 rows x 4 segs (2 per thread)
  const int arow = tid >> 2;
  const int aseg = (tid & 3) * 8;                  // element offset: 0,8,16,24
  const unsigned long long gA =
      (unsigned long long)(const void*)(xn + ((size_t)b * T_ + t0 + arow) * C_ + aseg);
  const unsigned long long gB0 =
      (unsigned long long)(const void*)(xn + ((size_t)b * T_ + c0 + arow) * C_ + aseg);
  const unsigned long long gB1 =
      (unsigned long long)(const void*)(xn + ((size_t)b * T_ + c0 + 64 + arow) * C_ + aseg);

  unsigned aoff[2], boff0[2], boff1[2];
#pragma unroll
  for (int u = 0; u < 2; ++u) {
    aoff[u]  = (unsigned)(unsigned long long)(void*)&lds_a[u][arow * 40 + aseg];
    boff0[u] = (unsigned)(unsigned long long)(void*)&lds_b[u][arow * 40 + aseg];
    boff1[u] = (unsigned)(unsigned long long)(void*)&lds_b[u][(64 + arow) * 40 + aseg];
  }

  v8f acc[4];
#pragma unroll
  for (int s = 0; s < 4; ++s) acc[s] = (v8f){0, 0, 0, 0, 0, 0, 0, 0};

  // preload K-step 0 into buffer 0
  async_ld_b128(aoff[0],  gA);
  async_ld_b128(boff0[0], gB0);
  async_ld_b128(boff1[0], gB1);
  wait_async0();
  __syncthreads();

  int cur = 0;
  for (int kk = 0; kk < C_; kk += 32) {
    if (kk + 32 < C_) {                            // prefetch next K-step into other buffer
      const unsigned long long koff = (unsigned long long)(kk + 32) * 2;
      async_ld_b128(aoff[cur ^ 1],  gA  + koff);
      async_ld_b128(boff0[cur ^ 1], gB0 + koff);
      async_ld_b128(boff1[cur ^ 1], gB1 + koff);
    }

    // A fragment: lane row M=m; elems 0..7 -> K = hi*8.., elems 8..15 -> K = 16+hi*8..
    FragAB af;
    const int arowm = (wrow * 16 + m) * 40;
    af.h[0] = *(const v8bf*)&lds_a[cur][arowm + hi * 8];
    af.h[1] = *(const v8bf*)&lds_a[cur][arowm + 16 + hi * 8];

#pragma unroll
    for (int sub = 0; sub < 4; ++sub) {            // B fragment: lane col N=m; K same as A (hi*8.., 16+hi*8..)
      FragAB bfr;
      const int nrow = (wcol * 64 + sub * 16 + m) * 40;
      bfr.h[0] = *(const v8bf*)&lds_b[cur][nrow + hi * 8];
      bfr.h[1] = *(const v8bf*)&lds_b[cur][nrow + 16 + hi * 8];
      acc[sub] = __builtin_amdgcn_wmma_f32_16x16x32_bf16(false, af.v, false, bfr.v,
                                                         (short)0, acc[sub], false, false);
    }

    wait_async0();
    __syncthreads();
    cur ^= 1;
  }

  // score = s>MARGIN ? s-MARGIN : -std[row t], reduce over the 16 rows of the wave strip
#pragma unroll
  for (int sub = 0; sub < 4; ++sub) {
    float cs = 0.f;
#pragma unroll
    for (int v = 0; v < 8; ++v) {
      const float sd = lds_std[wrow * 16 + v + hi * 8];
      const float s = acc[sub][v];
      cs += (s > MARGIN_) ? (s - MARGIN_) : (-sd);
    }
    cs += __shfl_down(cs, 16);
    if (hi == 0) atomicAdd(&lds_col[wcol * 64 + sub * 16 + m], cs);
  }
  __syncthreads();
  if (tid < 128) atomicAdd(&col_sum[(size_t)b * T_ + c0 + tid], lds_col[tid]);
}

// ---------------------------------------------------------------- per-batch selection
__global__ void __launch_bounds__(1024) k_sort_select(const float* __restrict__ col_sum,
                                                      int* __restrict__ a_idx,
                                                      int* __restrict__ b_idx,
                                                      int* __restrict__ kept_idx) {
  const int b   = blockIdx.x;
  const int tid = threadIdx.x;
  __shared__ unsigned long long s_key[T_];
  __shared__ int s_top[TWO_R_];

  for (int i = tid; i < T_; i += 1024) {
    unsigned u = __float_as_uint(col_sum[(size_t)b * T_ + i]);
    u = (u & 0x80000000u) ? ~u : (u ^ 0x80000000u);   // monotone ascending map
    s_key[i] = ((unsigned long long)(~u) << 32) | (unsigned)i;  // asc sort => desc score
  }
  __syncthreads();

  for (int k = 2; k <= T_; k <<= 1) {
    for (int j = k >> 1; j > 0; j >>= 1) {
      for (int t = tid; t < T_; t += 1024) {
        const int l = t ^ j;
        if (l > t) {
          const bool asc = ((t & k) == 0);
          const unsigned long long ka = s_key[t], kb = s_key[l];
          if ((ka > kb) == asc) { s_key[t] = kb; s_key[l] = ka; }
        }
      }
      __syncthreads();
    }
  }

  for (int i = tid; i < TWO_R_; i += 1024) {
    const int idx = (int)(s_key[i] & 0xFFFFFFFFull);
    s_top[i] = idx;
    if ((i & 1) == 0) a_idx[b * R_ + (i >> 1)] = idx;
    else              b_idx[b * R_ + (i >> 1)] = idx;
  }
  __syncthreads();

  int* flags = (int*)s_key;                 // repurpose LDS
  for (int i = tid; i < T_; i += 1024) flags[i] = 1;
  __syncthreads();
  if (tid < TWO_R_) flags[s_top[tid]] = 0;
  __syncthreads();
  if (tid == 0) {                           // stable compaction (ascending index)
    int c = 0;
    for (int t = 0; t < T_; ++t)
      if (flags[t]) kept_idx[b * KEEP_ + c++] = t;
  }
}

// ---------------------------------------------------------------- pair scores (r x r) via WMMA
__global__ void __launch_bounds__(32) k_pair(const unsigned short* __restrict__ xn,
                                             const int* __restrict__ a_idx,
                                             const int* __restrict__ b_idx,
                                             float* __restrict__ pair) {
  const int b  = blockIdx.z;
  const int i0 = blockIdx.y * 16;
  const int j0 = blockIdx.x * 16;
  const int lane = threadIdx.x & 31;
  const int m  = lane & 15;
  const int hi = lane >> 4;

  int ia = i0 + m; if (ia >= R_) ia = R_ - 1;
  int jb = j0 + m; if (jb >= R_) jb = R_ - 1;
  const int ta = a_idx[b * R_ + ia];
  const int tb = b_idx[b * R_ + jb];
  const unsigned short* pa = xn + ((size_t)b * T_ + ta) * C_;
  const unsigned short* pb = xn + ((size_t)b * T_ + tb) * C_;

  v8f acc = {0, 0, 0, 0, 0, 0, 0, 0};
  for (int kk = 0; kk < C_; kk += 32) {
    FragAB af, bfr;
    af.h[0]  = *(const v8bf*)&pa[kk + hi * 8];
    af.h[1]  = *(const v8bf*)&pa[kk + 16 + hi * 8];
    bfr.h[0] = *(const v8bf*)&pb[kk + hi * 8];
    bfr.h[1] = *(const v8bf*)&pb[kk + 16 + hi * 8];
    acc = __builtin_amdgcn_wmma_f32_16x16x32_bf16(false, af.v, false, bfr.v,
                                                  (short)0, acc, false, false);
  }
#pragma unroll
  for (int v = 0; v < 8; ++v) {
    const int i = i0 + v + hi * 8;
    const int j = j0 + m;
    if (i < R_ && j < R_) pair[((size_t)b * R_ + i) * R_ + j] = acc[v];
  }
}

// ---------------------------------------------------------------- argmax over j (first max wins)
__global__ void __launch_bounds__(128) k_argmax(const float* __restrict__ pair,
                                                int* __restrict__ dst_idx) {
  const int b = blockIdx.x;
  const int i = threadIdx.x;
  if (i >= R_) return;
  const float* prow = pair + ((size_t)b * R_ + i) * R_;
  float best = prow[0];
  int bi = 0;
  for (int j = 1; j < R_; ++j) {
    const float v = prow[j];
    if (v > best) { best = v; bi = j; }
  }
  dst_idx[b * R_ + i] = bi;
}

// ---------------------------------------------------------------- kept rows -> out
__global__ void __launch_bounds__(192) k_copy_ori(const float* __restrict__ x,
                                                  const int* __restrict__ kept_idx,
                                                  float* __restrict__ out) {
  const int k = blockIdx.x, b = blockIdx.y;
  const int t = kept_idx[b * KEEP_ + k];
  const float4* src = (const float4*)(x + ((size_t)b * T_ + t) * C_);
  float4* dst = (float4*)(out + ((size_t)b * OUT_T_ + k) * C_);
  dst[threadIdx.x] = src[threadIdx.x];
}

// ---------------------------------------------------------------- merged rows -> out
__global__ void __launch_bounds__(256) k_merge_dst(const float* __restrict__ x,
                                                   const int* __restrict__ a_idx,
                                                   const int* __restrict__ b_idx,
                                                   const int* __restrict__ dst_idx,
                                                   float* __restrict__ out) {
  const int j = blockIdx.x, b = blockIdx.y;
  const int tid = threadIdx.x;
  __shared__ int s_di[R_], s_ai[R_];
  if (tid < R_) { s_di[tid] = dst_idx[b * R_ + tid]; s_ai[tid] = a_idx[b * R_ + tid]; }
  __syncthreads();
  const int tb = b_idx[b * R_ + j];
  const float* drow = x + ((size_t)b * T_ + tb) * C_;
  float a0 = drow[tid], a1 = drow[tid + 256], a2 = drow[tid + 512];
  float cnt = 0.f;
  for (int i = 0; i < R_; ++i) {
    if (s_di[i] == j) {
      cnt += 1.f;
      const float* srow = x + ((size_t)b * T_ + s_ai[i]) * C_;
      a0 += srow[tid]; a1 += srow[tid + 256]; a2 += srow[tid + 512];
    }
  }
  const float w = 1.f / (1.f + cnt);
  float* orow = out + ((size_t)b * OUT_T_ + KEEP_ + j) * C_;
  orow[tid] = a0 * w; orow[tid + 256] = a1 * w; orow[tid + 512] = a2 * w;
}

// ----------------------------------------------------------------
extern "C" void kernel_launch(void* const* d_in, const int* in_sizes, int n_in,
                              void* d_out, int out_size, void* d_ws, size_t ws_size,
                              hipStream_t stream) {
  (void)in_sizes; (void)n_in; (void)out_size; (void)ws_size;
  const float* x = (const float*)d_in[0];
  float* out = (float*)d_out;

  char* ws = (char*)d_ws;
  size_t off = 0;
  auto take = [&](size_t bytes) -> char* {
    char* p = ws + off;
    off += (bytes + 255) & ~(size_t)255;
    return p;
  };
  unsigned short* xn = (unsigned short*)take((size_t)B_ * T_ * C_ * sizeof(unsigned short));
  float* stdv        = (float*)take((size_t)B_ * T_ * sizeof(float));
  float* col_sum     = (float*)take((size_t)B_ * T_ * sizeof(float));
  int*   a_idx       = (int*)take((size_t)B_ * R_ * sizeof(int));
  int*   b_idx       = (int*)take((size_t)B_ * R_ * sizeof(int));
  int*   kept_idx    = (int*)take((size_t)B_ * KEEP_ * sizeof(int));
  int*   dst_idx     = (int*)take((size_t)B_ * R_ * sizeof(int));
  float* pair        = (float*)take((size_t)B_ * R_ * R_ * sizeof(float));

  k_zero<<<(B_ * T_ + 255) / 256, 256, 0, stream>>>(col_sum, B_ * T_);
  k_normalize<<<B_ * T_, 256, 0, stream>>>(x, xn, stdv);
  k_gram_colsum<<<dim3(T_ / 128, T_ / 64, B_), 256, 0, stream>>>(xn, stdv, col_sum);
  k_sort_select<<<B_, 1024, 0, stream>>>(col_sum, a_idx, b_idx, kept_idx);
  k_pair<<<dim3((R_ + 15) / 16, (R_ + 15) / 16, B_), 32, 0, stream>>>(xn, a_idx, b_idx, pair);
  k_argmax<<<B_, 128, 0, stream>>>(pair, dst_idx);
  k_copy_ori<<<dim3(KEEP_, B_), 192, 0, stream>>>(x, kept_idx, out);
  k_merge_dst<<<dim3(R_, B_), 256, 0, stream>>>(x, a_idx, b_idx, dst_idx, out);
}
